// Multi_Branch_Attention_1125281431691
// MI455X (gfx1250) — compile-verified, matrix operand order checked by reading, not run on hardware
//
#include <hip/hip_runtime.h>

typedef __attribute__((ext_vector_type(16))) _Float16 v16h;
typedef __attribute__((ext_vector_type(8)))  _Float16 v8h;
typedef __attribute__((ext_vector_type(8)))  float    v8f;

#define HEADS 8
#define DHEAD 64
#define SEQ   2048
#define BATCH 4
#define DIM   512
#define INNER 512   // HEADS * DHEAD

// Load a 16x32 f16 WMMA A/B fragment from a row-major matrix (stride in halves).
// Lane l: row = row0 + (l&15); k elements = k0 + 8*(l>>4) + {0..7} and +16..23.
__device__ __forceinline__ v16h load_frag(const _Float16* __restrict__ base,
                                          int row0, int k0, int stride) {
  int lane = threadIdx.x & 31;
  const _Float16* p = base + (size_t)(row0 + (lane & 15)) * stride + k0 + ((lane >> 4) << 3);
  v8h lo = *reinterpret_cast<const v8h*>(p);
  v8h hi = *reinterpret_cast<const v8h*>(p + 16);
  v16h f;
#pragma unroll
  for (int i = 0; i < 8; ++i) { f[i] = lo[i]; f[i + 8] = hi[i]; }
  return f;
}

__global__ void cvt_f32_f16(const float* __restrict__ in, _Float16* __restrict__ out, int n) {
  int i = blockIdx.x * 256 + threadIdx.x;
  if (i < n) out[i] = (_Float16)in[i];
}

// Wt[c*K + k] = W[k*C + c]  (f32 -> f16 transpose)
__global__ void wtrans(const float* __restrict__ W, _Float16* __restrict__ Wt, int K, int C) {
  int i = blockIdx.x * 256 + threadIdx.x;
  if (i < K * C) {
    int c = i % C, k = i / C;
    Wt[(size_t)c * K + k] = (_Float16)W[i];
  }
}

// GEMM: [8192 x 512] (f16) x Wt^T -> scatter into per-head layouts.
// col < 512  -> dstK[((b*H+h)*SEQ + n)*64 + d] = val*scale   (Q or K, [B,H,N,64])
// col >= 512 -> dstV[((b*H+h)*64 + d)*SEQ + n] = val         (V transposed, [B,H,64,N])
__global__ __launch_bounds__(256) void proj_gemm(const _Float16* __restrict__ X,
                                                 const _Float16* __restrict__ Wt,
                                                 _Float16* __restrict__ dstK,
                                                 _Float16* __restrict__ dstV,
                                                 int ncols, float scale) {
  int lane = threadIdx.x & 31;
  int wid  = threadIdx.x >> 5;
  int id   = blockIdx.x * 8 + wid;
  const int MT = (BATCH * SEQ) / 16;  // 512 row tiles
  int m0 = (id % MT) * 16;
  int c0 = (id / MT) * 64;
  if (c0 >= ncols) return;

  v8f acc0 = {}, acc1 = {}, acc2 = {}, acc3 = {};
#pragma unroll 1
  for (int k0 = 0; k0 < DIM; k0 += 32) {
    v16h a  = load_frag(X,  m0,      k0, DIM);
    v16h b0 = load_frag(Wt, c0,      k0, DIM);
    v16h b1 = load_frag(Wt, c0 + 16, k0, DIM);
    v16h b2 = load_frag(Wt, c0 + 32, k0, DIM);
    v16h b3 = load_frag(Wt, c0 + 48, k0, DIM);
    acc0 = __builtin_amdgcn_wmma_f32_16x16x32_f16(false, a, false, b0, (short)0, acc0, false, false);
    acc1 = __builtin_amdgcn_wmma_f32_16x16x32_f16(false, a, false, b1, (short)0, acc1, false, false);
    acc2 = __builtin_amdgcn_wmma_f32_16x16x32_f16(false, a, false, b2, (short)0, acc2, false, false);
    acc3 = __builtin_amdgcn_wmma_f32_16x16x32_f16(false, a, false, b3, (short)0, acc3, false, false);
  }

  v8f accs[4] = {acc0, acc1, acc2, acc3};
  int rowb = m0 + ((lane >> 4) << 3);
  int cl   = lane & 15;
#pragma unroll
  for (int t = 0; t < 4; ++t) {
#pragma unroll
    for (int r = 0; r < 8; ++r) {
      float val = accs[t][r];
      int row = rowb + r;
      int col = c0 + t * 16 + cl;
      int b = row >> 11, n = row & (SEQ - 1);
      if (col < INNER) {
        int h = col >> 6, d = col & 63;
        dstK[(((size_t)(b * HEADS + h)) * SEQ + n) * DHEAD + d] = (_Float16)(val * scale);
      } else {
        int cc = col - INNER;
        int h = cc >> 6, d = cc & 63;
        dstV[(((size_t)(b * HEADS + h)) * DHEAD + d) * SEQ + n] = (_Float16)val;
      }
    }
  }
}

// Issue the per-thread share of one 32-key K/V block copy into LDS.
// K block: 32 rows x 64 halves (stride DHEAD);  V block: 64 rows x 32 halves (from stride SEQ).
// 256 threads x 16B each per tensor -> each WAVE issues exactly 2 async instructions.
__device__ __forceinline__ void issue_async_copies(const _Float16* __restrict__ Kg,
                                                   const _Float16* __restrict__ Vg,
                                                   _Float16* kdst, _Float16* vdst, int j) {
  int t = threadIdx.x;
  const _Float16* gk = Kg + (size_t)(j + (t >> 3)) * DHEAD + (t & 7) * 8;
  const _Float16* gv = Vg + (size_t)(t >> 2) * SEQ + j + (t & 3) * 8;
  uint32_t lk = (uint32_t)(uintptr_t)(kdst + (t >> 3) * DHEAD + (t & 7) * 8);
  uint32_t lv = (uint32_t)(uintptr_t)(vdst + (t >> 2) * 32 + (t & 3) * 8);
  asm volatile("global_load_async_to_lds_b128 %0, %1, off" :: "v"(lk), "v"(gk) : "memory");
  asm volatile("global_load_async_to_lds_b128 %0, %1, off" :: "v"(lv), "v"(gv) : "memory");
}

// Flash attention: block = one (branch,b,h), 8 waves x 16 query rows = 128 rows.
// K/V blocks staged once per block into LDS via async copies (double buffered),
// shared by all 8 waves -> 8x less L2 traffic than per-wave streaming.
__global__ __launch_bounds__(256) void flash_attn(const _Float16* __restrict__ Q1,
                                                  const _Float16* __restrict__ Q2,
                                                  const _Float16* __restrict__ Kt,
                                                  const _Float16* __restrict__ Vt,
                                                  float* __restrict__ out) {
  // [K0 2048][K1 2048][V0 2048][V1 2048][P 8*512]  halves = 24KB
  __shared__ __align__(16) _Float16 smem[12288];
  int lane = threadIdx.x & 31;
  int wid  = threadIdx.x >> 5;
  int bid  = blockIdx.x;
  int bhz  = bid & 63;       // (branch, b, h)
  int qc   = bid >> 6;       // 0..15
  int q0   = qc * 128 + wid * 16;
  int h    = bhz & 7;
  int b    = (bhz >> 3) & 3;
  int br   = bhz >> 5;

  const _Float16* Q  = (br ? Q2 : Q1) + ((size_t)(b * HEADS + h)) * SEQ * DHEAD;
  const _Float16* Kg = Kt + ((size_t)(b * HEADS + h)) * SEQ * DHEAD;
  const _Float16* Vg = Vt + ((size_t)(b * HEADS + h)) * DHEAD * SEQ;
  float* O = out + (size_t)br * ((size_t)BATCH * SEQ * INNER);

  v16h qa0 = load_frag(Q, q0, 0,  DHEAD);
  v16h qa1 = load_frag(Q, q0, 32, DHEAD);

  v8f acc[4] = {};
  float mrow[8], lsum[8];
#pragma unroll
  for (int r = 0; r < 8; ++r) { mrow[r] = -1e30f; lsum[r] = 0.f; }

  _Float16* lp = smem + 8192 + wid * 512;  // per-wave P transpose slice
  int rb = (lane >> 4) << 3;
  int cl = lane & 15;

  issue_async_copies(Kg, Vg, smem, smem + 4096, 0);  // prologue: block 0 -> buffer 0

#pragma unroll 1
  for (int jb = 0; jb < SEQ / 32; ++jb) {
    int cur = jb & 1;
    const _Float16* kcur = smem + cur * 2048;
    const _Float16* vcur = smem + 4096 + cur * 2048;
    if (jb < SEQ / 32 - 1) {
      issue_async_copies(Kg, Vg, smem + (1 - cur) * 2048, smem + 4096 + (1 - cur) * 2048,
                         (jb + 1) * 32);
      asm volatile("s_wait_asynccnt 2" ::: "memory");   // current buffer's 2 copies done
    } else {
      asm volatile("s_wait_asynccnt 0" ::: "memory");
    }
    __syncthreads();  // all waves' copies landed

    v16h kb00 = load_frag(kcur, 0,  0,  DHEAD);
    v16h kb01 = load_frag(kcur, 0,  32, DHEAD);
    v16h kb10 = load_frag(kcur, 16, 0,  DHEAD);
    v16h kb11 = load_frag(kcur, 16, 32, DHEAD);
    v8f s0 = {}, s1 = {};
    s0 = __builtin_amdgcn_wmma_f32_16x16x32_f16(false, qa0, false, kb00, (short)0, s0, false, false);
    s0 = __builtin_amdgcn_wmma_f32_16x16x32_f16(false, qa1, false, kb01, (short)0, s0, false, false);
    s1 = __builtin_amdgcn_wmma_f32_16x16x32_f16(false, qa0, false, kb10, (short)0, s1, false, false);
    s1 = __builtin_amdgcn_wmma_f32_16x16x32_f16(false, qa1, false, kb11, (short)0, s1, false, false);

    // Online softmax. Lane l holds rows rb+r, cols cl (s0) / 16+cl (s1).
    // A row's 16 columns live in one 16-lane half -> xor masks 1,2,4,8 reduce it.
#pragma unroll
    for (int r = 0; r < 8; ++r) {
      float mx = fmaxf(s0[r], s1[r]);
      mx = fmaxf(mx, __shfl_xor(mx, 1));
      mx = fmaxf(mx, __shfl_xor(mx, 2));
      mx = fmaxf(mx, __shfl_xor(mx, 4));
      mx = fmaxf(mx, __shfl_xor(mx, 8));
      float mnew  = fmaxf(mrow[r], mx);
      float alpha = __expf(mrow[r] - mnew);
      float p0 = __expf(s0[r] - mnew);
      float p1 = __expf(s1[r] - mnew);
      float rs = p0 + p1;
      rs += __shfl_xor(rs, 1);
      rs += __shfl_xor(rs, 2);
      rs += __shfl_xor(rs, 4);
      rs += __shfl_xor(rs, 8);
      lsum[r] = lsum[r] * alpha + rs;
      mrow[r] = mnew;
      acc[0][r] *= alpha; acc[1][r] *= alpha; acc[2][r] *= alpha; acc[3][r] *= alpha;
      lp[(rb + r) * 32 + cl]      = (_Float16)p0;
      lp[(rb + r) * 32 + 16 + cl] = (_Float16)p1;
    }

    // P: D-layout -> A-fragment layout via LDS (same-wave DS ops are in-order).
    asm volatile("s_wait_dscnt 0" ::: "memory");
    const _Float16* rp = lp + cl * 32 + rb;
    v8h lo = *reinterpret_cast<const v8h*>(rp);
    v8h hi = *reinterpret_cast<const v8h*>(rp + 16);
    v16h pa;
#pragma unroll
    for (int i = 0; i < 8; ++i) { pa[i] = lo[i]; pa[i + 8] = hi[i]; }

    v16h vb0 = load_frag(vcur, 0,  0, 32);
    v16h vb1 = load_frag(vcur, 16, 0, 32);
    v16h vb2 = load_frag(vcur, 32, 0, 32);
    v16h vb3 = load_frag(vcur, 48, 0, 32);
    acc[0] = __builtin_amdgcn_wmma_f32_16x16x32_f16(false, pa, false, vb0, (short)0, acc[0], false, false);
    acc[1] = __builtin_amdgcn_wmma_f32_16x16x32_f16(false, pa, false, vb1, (short)0, acc[1], false, false);
    acc[2] = __builtin_amdgcn_wmma_f32_16x16x32_f16(false, pa, false, vb2, (short)0, acc[2], false, false);
    acc[3] = __builtin_amdgcn_wmma_f32_16x16x32_f16(false, pa, false, vb3, (short)0, acc[3], false, false);

    __syncthreads();  // everyone done reading this buffer before it is refilled
  }

#pragma unroll
  for (int t = 0; t < 4; ++t) {
#pragma unroll
    for (int r = 0; r < 8; ++r) {
      int n = q0 + rb + r;
      int d = h * DHEAD + t * 16 + cl;
      O[((size_t)(b * SEQ + n)) * INNER + d] = acc[t][r] / lsum[r];
    }
  }
}

extern "C" void kernel_launch(void* const* d_in, const int* in_sizes, int n_in,
                              void* d_out, int out_size, void* d_ws, size_t ws_size,
                              hipStream_t stream) {
  const float* q1  = (const float*)d_in[0];
  const float* q2  = (const float*)d_in[1];
  const float* xkv = (const float*)d_in[2];
  const float* Wq1 = (const float*)d_in[3];
  const float* Wq2 = (const float*)d_in[4];
  const float* Wkv = (const float*)d_in[5];
  float* out = (float*)d_out;

  const size_t NBH = (size_t)BATCH * SEQ * INNER;  // 4,194,304 elements
  _Float16* ws  = (_Float16*)d_ws;
  _Float16* Q1h = ws;
  _Float16* Q2h = Q1h + NBH;
  _Float16* Kh  = Q2h + NBH;
  _Float16* Vth = Kh  + NBH;
  _Float16* X1h = Vth + NBH;
  _Float16* X2h = X1h + NBH;
  _Float16* Xkh = X2h + NBH;
  _Float16* Wt1 = Xkh + NBH;
  _Float16* Wt2 = Wt1 + (size_t)DIM * INNER;
  _Float16* Wtk = Wt2 + (size_t)DIM * INNER;

  int nx = (int)NBH;
  int cb = (nx + 255) / 256;
  cvt_f32_f16<<<cb, 256, 0, stream>>>(q1,  X1h, nx);
  cvt_f32_f16<<<cb, 256, 0, stream>>>(q2,  X2h, nx);
  cvt_f32_f16<<<cb, 256, 0, stream>>>(xkv, Xkh, nx);
  wtrans<<<(DIM * INNER + 255) / 256, 256, 0, stream>>>(Wq1, Wt1, DIM, INNER);
  wtrans<<<(DIM * INNER + 255) / 256, 256, 0, stream>>>(Wq2, Wt2, DIM, INNER);
  wtrans<<<(DIM * 2 * INNER + 255) / 256, 256, 0, stream>>>(Wkv, Wtk, DIM, 2 * INNER);

  const float scale = 0.125f;  // DHEAD^-0.5
  const int MT = (BATCH * SEQ) / 16;  // 512
  proj_gemm<<<(MT * (INNER / 64)) / 8,     256, 0, stream>>>(X1h, Wt1, Q1h, nullptr, INNER,     scale);
  proj_gemm<<<(MT * (INNER / 64)) / 8,     256, 0, stream>>>(X2h, Wt2, Q2h, nullptr, INNER,     scale);
  proj_gemm<<<(MT * (2 * INNER / 64)) / 8, 256, 0, stream>>>(Xkh, Wtk, Kh,  Vth,     2 * INNER, 1.0f);

  // 64 (branch,b,h) x 16 query chunks = 1024 blocks of 8 waves
  flash_attn<<<1024, 256, 0, stream>>>(Q1h, Q2h, Kh, Vth, out);
}
